// BigramLanguageModel_30313879175395
// MI455X (gfx1250) — matrix operand order repaired and checked by reading, not compiled, not run on hardware
//
#include <hip/hip_runtime.h>
#include <hip/hip_bf16.h>
#include <math.h>

// Problem dims (fixed by reference)
#define BB 4
#define TT 1024
#define EE 1024
#define HH 16
#define HSD 64
#define LL 8
#define VV 32000
#define FFD 4096
#define MROWS (BB*TT)   // 4096 token rows

typedef __bf16 bf16;
typedef __attribute__((ext_vector_type(16))) __bf16 v16bf;
typedef __attribute__((ext_vector_type(8)))  float  v8f;

union U32x8 { uint4 u[2]; v16bf v; };   // 32B: one WMMA 16-bit operand fragment per lane
union U8BF  { uint4 u; bf16 e[8]; };

static __device__ __forceinline__ v8f wmma_bf16(v16bf a, v16bf b, v8f c) {
  return __builtin_amdgcn_wmma_f32_16x16x32_bf16(false, a, false, b, (short)0, c, false, false);
}

// CDNA5 async global->LDS copy (16B per lane, ASYNCcnt-tracked)
static __device__ __forceinline__ void async_ld_b128(const bf16* lds_dst, const bf16* gsrc) {
  unsigned int loff = (unsigned int)(unsigned long long)(__UINTPTR_TYPE__)lds_dst;
  unsigned long long g = (unsigned long long)(__UINTPTR_TYPE__)gsrc;
  asm volatile("global_load_async_to_lds_b128 %0, %1, off"
               :: "v"(loff), "v"(g) : "memory");
}
static __device__ __forceinline__ void wait_async0() {
#if __has_builtin(__builtin_amdgcn_s_wait_asynccnt)
  __builtin_amdgcn_s_wait_asynccnt(0);
#else
  asm volatile("s_wait_asynccnt 0" ::: "memory");
#endif
}

// ---------------------------------------------------------------- convert f32 -> bf16
__global__ void k_cvt_bf16(const float* __restrict__ in, bf16* __restrict__ out, long n) {
  long i = (long)blockIdx.x * blockDim.x + threadIdx.x;
  long st = (long)gridDim.x * blockDim.x;
  for (; i < n; i += st) out[i] = (bf16)in[i];
}

// ---------------------------------------------------------------- embedding gather
__global__ void k_embed(const int* __restrict__ idx, const float* __restrict__ tok,
                        const float* __restrict__ pos, float* __restrict__ x) {
  int e  = blockIdx.x * blockDim.x + threadIdx.x;   // grid.x = EE/256
  int bt = blockIdx.y;                               // 0..MROWS-1
  int t  = bt & (TT - 1);
  int tk = idx[bt];
  x[(size_t)bt * EE + e] = tok[(size_t)tk * EE + e] + pos[(size_t)t * EE + e];
}

// ---------------------------------------------------------------- layernorm (f32 in, bf16 out)
__global__ __launch_bounds__(256)
void k_ln(const float* __restrict__ x, const float* __restrict__ s,
          const float* __restrict__ b, bf16* __restrict__ out) {
  __shared__ float red[16];
  int row = blockIdx.x;
  const float* xr = x + (size_t)row * EE;
  int base = threadIdx.x * 4;                        // EE/256 == 4
  float4 v = *(const float4*)&xr[base];
  float sum = v.x + v.y + v.z + v.w;
  float sq  = v.x*v.x + v.y*v.y + v.z*v.z + v.w*v.w;
  #pragma unroll
  for (int m = 1; m < 32; m <<= 1) { sum += __shfl_xor(sum, m, 32); sq += __shfl_xor(sq, m, 32); }
  int wid = threadIdx.x >> 5;
  if ((threadIdx.x & 31) == 0) { red[wid] = sum; red[wid + 8] = sq; }
  __syncthreads();
  float ts = 0.f, tq = 0.f;
  #pragma unroll
  for (int i = 0; i < 8; ++i) { ts += red[i]; tq += red[i + 8]; }
  float mean = ts * (1.0f / EE);
  float var  = tq * (1.0f / EE) - mean * mean;
  float rstd = rsqrtf(var + 1e-5f);
  bf16* orow = out + (size_t)row * EE;
  float xe[4] = {v.x, v.y, v.z, v.w};
  #pragma unroll
  for (int j = 0; j < 4; ++j) {
    int e = base + j;
    orow[e] = (bf16)((xe[j] - mean) * rstd * s[e] + b[e]);
  }
}

// ---------------------------------------------------------------- tiled WMMA GEMM
// C[M,N] = A[M,K](bf16,row) @ B[K,N](bf16,row); block tile 128x128, K-step 64.
// EP: 0 = bf16 store; 1 = bias+relu -> bf16; 2 = bias+residual -> f32; 3 = bias -> f32
#define LPAD 72
template<int EP>
__global__ __launch_bounds__(256)
void k_gemm(const bf16* __restrict__ A, const bf16* __restrict__ Bm,
            const float* __restrict__ bias, const float* __restrict__ resid,
            float* __restrict__ outF, bf16* __restrict__ outB, int N, int K) {
  __shared__ __align__(16) bf16 As[128 * LPAD];   // [row][k] pad-72
  __shared__ __align__(16) bf16 Bt[128 * LPAD];   // [col][k] (B transposed) pad-72
  int tid  = threadIdx.x;
  int lane = tid & 31, wid = tid >> 5;
  int half = lane >> 4, l16 = lane & 15;
  int wm = wid & 3, wn = wid >> 2;               // 4 x 2 wave grid, wave tile 32x64
  int m0 = blockIdx.y * 128, n0 = blockIdx.x * 128;

  v8f acc[2][4];
  #pragma unroll
  for (int i = 0; i < 2; ++i)
    #pragma unroll
    for (int j = 0; j < 4; ++j) acc[i][j] = (v8f){0.f,0.f,0.f,0.f,0.f,0.f,0.f,0.f};

  int nk = K >> 6;
  for (int kt = 0; kt < nk; ++kt) {
    int k0 = kt << 6;
    __syncthreads();
    // stage A tile 128x64 via async global->LDS (4 x 16B per thread)
    #pragma unroll
    for (int i = 0; i < 4; ++i) {
      int vd = tid + i * 256;
      int r = vd >> 3, cv = (vd & 7) << 3;
      async_ld_b128(&As[r * LPAD + cv], &A[(size_t)(m0 + r) * K + k0 + cv]);
    }
    // stage B tile 64x128 transposed (register round-trip needed)
    #pragma unroll
    for (int i = 0; i < 4; ++i) {
      int vd = tid + i * 256;
      int r = vd >> 4, cv = (vd & 15) << 3;
      U8BF u; u.u = *(const uint4*)&Bm[(size_t)(k0 + r) * N + n0 + cv];
      #pragma unroll
      for (int j = 0; j < 8; ++j) Bt[(cv + j) * LPAD + r] = u.e[j];
    }
    // prefetch next K-tile
    if (kt + 1 < nk) {
      __builtin_prefetch(&A[(size_t)(m0 + (tid >> 1)) * K + k0 + 64], 0, 0);
      __builtin_prefetch(&Bm[(size_t)(k0 + 64 + (tid >> 3)) * N + n0], 0, 0);
    }
    wait_async0();
    __syncthreads();
    #pragma unroll
    for (int s = 0; s < 2; ++s) {       // two 32-deep WMMA K-steps per stage
      v16bf af[2], bfm[4];
      #pragma unroll
      for (int fm = 0; fm < 2; ++fm) {
        U32x8 u; int row = wm * 32 + fm * 16 + l16; int kb = s * 32 + half * 8;
        u.u[0] = *(const uint4*)&As[row * LPAD + kb];
        u.u[1] = *(const uint4*)&As[row * LPAD + kb + 16];
        af[fm] = u.v;
      }
      #pragma unroll
      for (int fn = 0; fn < 4; ++fn) {
        U32x8 u; int col = wn * 64 + fn * 16 + l16; int ks = s * 32 + half * 8;
        u.u[0] = *(const uint4*)&Bt[col * LPAD + ks];
        u.u[1] = *(const uint4*)&Bt[col * LPAD + ks + 16];
        bfm[fn] = u.v;
      }
      #pragma unroll
      for (int fm = 0; fm < 2; ++fm)
        #pragma unroll
        for (int fn = 0; fn < 4; ++fn)
          acc[fm][fn] = wmma_bf16(af[fm], bfm[fn], acc[fm][fn]);
    }
  }
  // epilogue
  #pragma unroll
  for (int fm = 0; fm < 2; ++fm)
    #pragma unroll
    for (int fn = 0; fn < 4; ++fn)
      #pragma unroll
      for (int r = 0; r < 8; ++r) {
        int row = m0 + wm * 32 + fm * 16 + half * 8 + r;
        int col = n0 + wn * 64 + fn * 16 + l16;
        size_t o = (size_t)row * N + col;
        float cv = acc[fm][fn][r];
        if (EP == 0) {
          outB[o] = (bf16)cv;
        } else if (EP == 1) {
          float t = cv + bias[col];
          outB[o] = (bf16)(t > 0.f ? t : 0.f);
        } else if (EP == 2) {
          outF[o] = resid[o] + cv + bias[col];
        } else {
          outF[o] = cv + bias[col];
        }
      }
}

// ---------------------------------------------------------------- flash attention (causal)
// grid: (T/128, B*H); block 256 = 8 waves; wave owns 16 query rows.
__global__ __launch_bounds__(256)
void k_attn(const bf16* __restrict__ q, const bf16* __restrict__ kx,
            const bf16* __restrict__ vx, bf16* __restrict__ o) {
  __shared__ __align__(16) bf16 Kl[32 * 72];       // [key][d]   pad-72
  __shared__ __align__(16) bf16 Vt[64 * 40];       // [d][key]   pad-40
  __shared__ __align__(16) bf16 Pl[8 * 16 * 40];   // per-wave P [row][key] pad-40
  int tid = threadIdx.x, lane = tid & 31, wid = tid >> 5;
  int half = lane >> 4, l16 = lane & 15;
  int qblk = blockIdx.x, bh = blockIdx.y;
  int bb = bh / HH, hh = bh % HH;
  size_t baseoff = (size_t)bb * TT * EE + (size_t)hh * HSD;

  // Q fragments (16x64 -> two 16x32 A-frags), loaded once
  int qrow = qblk * 128 + wid * 16 + l16;
  const bf16* qrp = q + baseoff + (size_t)qrow * EE;
  v16bf qa[2];
  #pragma unroll
  for (int st = 0; st < 2; ++st) {
    U32x8 u; int kb = st * 32 + half * 8;
    u.u[0] = *(const uint4*)&qrp[kb];
    u.u[1] = *(const uint4*)&qrp[kb + 16];
    qa[st] = u.v;
  }

  v8f of[4];
  #pragma unroll
  for (int n = 0; n < 4; ++n) of[n] = (v8f){0.f,0.f,0.f,0.f,0.f,0.f,0.f,0.f};
  float mr[8], lr[8];
  #pragma unroll
  for (int r = 0; r < 8; ++r) { mr[r] = -1e30f; lr[r] = 0.f; }

  bf16* pw = &Pl[wid * 640];
  int nk = (qblk + 1) * 4;            // 32-key tiles up to causal bound
  for (int kt = 0; kt < nk; ++kt) {
    int k0 = kt * 32;
    __syncthreads();
    { // stage K (verbatim copy -> async), V (transposed -> manual)
      int key = tid >> 3, dv = (tid & 7) << 3;
      async_ld_b128(&Kl[key * 72 + dv], &kx[baseoff + (size_t)(k0 + key) * EE + dv]);
      U8BF u; u.u = *(const uint4*)&vx[baseoff + (size_t)(k0 + key) * EE + dv];
      #pragma unroll
      for (int j = 0; j < 8; ++j) Vt[(dv + j) * 40 + key] = u.e[j];
    }
    wait_async0();
    __syncthreads();

    // S = Q @ K^T  (16 x 32 as two 16x16 C-frags), scale + causal mask
    float sc[2][8];
    #pragma unroll
    for (int c = 0; c < 2; ++c) {
      v8f s = (v8f){0.f,0.f,0.f,0.f,0.f,0.f,0.f,0.f};
      #pragma unroll
      for (int st = 0; st < 2; ++st) {
        U32x8 u; int col = c * 16 + l16; int ks = st * 32 + half * 8;
        u.u[0] = *(const uint4*)&Kl[col * 72 + ks];
        u.u[1] = *(const uint4*)&Kl[col * 72 + ks + 16];
        s = wmma_bf16(qa[st], u.v, s);
      }
      int kj = k0 + c * 16 + l16;
      #pragma unroll
      for (int r = 0; r < 8; ++r) {
        int qi = qblk * 128 + wid * 16 + half * 8 + r;
        float x = s[r] * 0.125f;          // HS^-0.5
        sc[c][r] = (kj > qi) ? -1e30f : x;
      }
    }
    // online softmax: row reductions across 16-lane halves
    float alpha[8], p0[8], p1[8];
    #pragma unroll
    for (int r = 0; r < 8; ++r) {
      float v = fmaxf(sc[0][r], sc[1][r]);
      v = fmaxf(v, __shfl_xor(v, 1, 32));
      v = fmaxf(v, __shfl_xor(v, 2, 32));
      v = fmaxf(v, __shfl_xor(v, 4, 32));
      v = fmaxf(v, __shfl_xor(v, 8, 32));
      float mn = fmaxf(mr[r], v);
      alpha[r] = __expf(mr[r] - mn);
      mr[r] = mn;
      p0[r] = __expf(sc[0][r] - mn);
      p1[r] = __expf(sc[1][r] - mn);
      float ss = p0[r] + p1[r];
      ss += __shfl_xor(ss, 1, 32);
      ss += __shfl_xor(ss, 2, 32);
      ss += __shfl_xor(ss, 4, 32);
      ss += __shfl_xor(ss, 8, 32);
      lr[r] = lr[r] * alpha[r] + ss;
    }
    // stash P (bf16) in per-wave LDS: C-layout -> memory
    #pragma unroll
    for (int r = 0; r < 8; ++r) {
      pw[(half * 8 + r) * 40 + l16]      = (bf16)p0[r];
      pw[(half * 8 + r) * 40 + 16 + l16] = (bf16)p1[r];
    }
    // rescale accumulated O
    #pragma unroll
    for (int n = 0; n < 4; ++n)
      #pragma unroll
      for (int r = 0; r < 8; ++r) of[n][r] *= alpha[r];
    // O += P @ V  (A-frag from LDS, B-frags from V^T)
    U32x8 up; int kb = half * 8;
    up.u[0] = *(const uint4*)&pw[l16 * 40 + kb];
    up.u[1] = *(const uint4*)&pw[l16 * 40 + kb + 16];
    v16bf pa = up.v;
    #pragma unroll
    for (int n = 0; n < 4; ++n) {
      U32x8 u; int col = n * 16 + l16; int ks = half * 8;
      u.u[0] = *(const uint4*)&Vt[col * 40 + ks];
      u.u[1] = *(const uint4*)&Vt[col * 40 + ks + 16];
      of[n] = wmma_bf16(pa, u.v, of[n]);
    }
  }
  // normalize and store
  #pragma unroll
  for (int r = 0; r < 8; ++r) {
    float inv = 1.0f / lr[r];
    int qi = qblk * 128 + wid * 16 + half * 8 + r;
    #pragma unroll
    for (int n = 0; n < 4; ++n)
      o[baseoff + (size_t)qi * EE + n * 16 + l16] = (bf16)(of[n][r] * inv);
  }
}

// ---------------------------------------------------------------- NLL loss
__global__ __launch_bounds__(256)
void k_loss_rows(const float* __restrict__ logits, const int* __restrict__ targets,
                 float* __restrict__ row_nll) {
  __shared__ float red[8];
  int row = blockIdx.x;
  const float* lr = logits + (size_t)row * VV;
  float mx = -1e30f;
  for (int i = threadIdx.x; i < VV; i += 256) mx = fmaxf(mx, lr[i]);
  #pragma unroll
  for (int m = 1; m < 32; m <<= 1) mx = fmaxf(mx, __shfl_xor(mx, m, 32));
  int wid = threadIdx.x >> 5;
  if ((threadIdx.x & 31) == 0) red[wid] = mx;
  __syncthreads();
  float bm = -1e30f;
  #pragma unroll
  for (int i = 0; i < 8; ++i) bm = fmaxf(bm, red[i]);
  __syncthreads();
  float se = 0.f;
  for (int i = threadIdx.x; i < VV; i += 256) se += __expf(lr[i] - bm);
  #pragma unroll
  for (int m = 1; m < 32; m <<= 1) se += __shfl_xor(se, m, 32);
  if ((threadIdx.x & 31) == 0) red[wid] = se;
  __syncthreads();
  if (threadIdx.x == 0) {
    float ts = 0.f;
    for (int i = 0; i < 8; ++i) ts += red[i];
    row_nll[row] = (bm + logf(ts)) - lr[targets[row]];
  }
}

__global__ __launch_bounds__(256)
void k_loss_reduce(const float* __restrict__ row_nll, float* __restrict__ out) {
  __shared__ float red[8];
  float s = 0.f;
  for (int i = threadIdx.x; i < MROWS; i += 256) s += row_nll[i];
  #pragma unroll
  for (int m = 1; m < 32; m <<= 1) s += __shfl_xor(s, m, 32);
  if ((threadIdx.x & 31) == 0) red[threadIdx.x >> 5] = s;
  __syncthreads();
  if (threadIdx.x == 0) {
    float t = 0.f;
    for (int i = 0; i < 8; ++i) t += red[i];
    out[0] = t / (float)MROWS;
  }
}

// ---------------------------------------------------------------- host driver
extern "C" void kernel_launch(void* const* d_in, const int* in_sizes, int n_in,
                              void* d_out, int out_size, void* d_ws, size_t ws_size,
                              hipStream_t stream) {
  (void)in_sizes; (void)n_in; (void)out_size; (void)ws_size;
  const int*   idx     = (const int*)d_in[0];
  const int*   targets = (const int*)d_in[1];
  const float* tok_emb = (const float*)d_in[2];
  const float* pos_emb = (const float*)d_in[3];
  const float* Wq  = (const float*)d_in[4];
  const float* Wk  = (const float*)d_in[5];
  const float* Wv  = (const float*)d_in[6];
  const float* Wo  = (const float*)d_in[7];
  const float* bo  = (const float*)d_in[8];
  const float* ln1s = (const float*)d_in[9];
  const float* ln1b = (const float*)d_in[10];
  const float* ln2s = (const float*)d_in[11];
  const float* ln2b = (const float*)d_in[12];
  const float* W1  = (const float*)d_in[13];
  const float* b1  = (const float*)d_in[14];
  const float* W2  = (const float*)d_in[15];
  const float* b2  = (const float*)d_in[16];
  const float* lnfs = (const float*)d_in[17];
  const float* lnfb = (const float*)d_in[18];
  const float* Wlm = (const float*)d_in[19];
  const float* blm = (const float*)d_in[20];

  float* logits = (float*)d_out;
  float* loss   = logits + (size_t)MROWS * VV;

  // workspace carve-out
  unsigned char* wsb = (unsigned char*)d_ws;
  size_t off = 0;
  auto take = [&](size_t bytes) -> void* {
    void* p = wsb + off;
    off += (bytes + 255) & ~(size_t)255;
    return p;
  };
  bf16* wq_b  = (bf16*)take(sizeof(bf16) * (size_t)LL * EE * EE);
  bf16* wk_b  = (bf16*)take(sizeof(bf16) * (size_t)LL * EE * EE);
  bf16* wv_b  = (bf16*)take(sizeof(bf16) * (size_t)LL * EE * EE);
  bf16* wo_b  = (bf16*)take(sizeof(bf16) * (size_t)LL * EE * EE);
  bf16* w1_b  = (bf16*)take(sizeof(bf16) * (size_t)LL * EE * FFD);
  bf16* w2_b  = (bf16*)take(sizeof(bf16) * (size_t)LL * FFD * EE);
  bf16* wlm_b = (bf16*)take(sizeof(bf16) * (size_t)EE * VV);
  float* x    = (float*)take(sizeof(float) * (size_t)MROWS * EE);
  bf16* hbuf  = (bf16*)take(sizeof(bf16) * (size_t)MROWS * EE);
  bf16* qbuf  = (bf16*)take(sizeof(bf16) * (size_t)MROWS * EE);
  bf16* kbuf  = (bf16*)take(sizeof(bf16) * (size_t)MROWS * EE);
  bf16* vbuf  = (bf16*)take(sizeof(bf16) * (size_t)MROWS * EE);
  bf16* obuf  = (bf16*)take(sizeof(bf16) * (size_t)MROWS * EE);
  bf16* abuf  = (bf16*)take(sizeof(bf16) * (size_t)MROWS * FFD);
  bf16* xfbuf = (bf16*)take(sizeof(bf16) * (size_t)MROWS * EE);
  float* rnll = (float*)take(sizeof(float) * MROWS);

  // weights -> bf16 (done every launch; deterministic)
  k_cvt_bf16<<<2048, 256, 0, stream>>>(Wq,  wq_b,  (long)LL * EE * EE);
  k_cvt_bf16<<<2048, 256, 0, stream>>>(Wk,  wk_b,  (long)LL * EE * EE);
  k_cvt_bf16<<<2048, 256, 0, stream>>>(Wv,  wv_b,  (long)LL * EE * EE);
  k_cvt_bf16<<<2048, 256, 0, stream>>>(Wo,  wo_b,  (long)LL * EE * EE);
  k_cvt_bf16<<<4096, 256, 0, stream>>>(W1,  w1_b,  (long)LL * EE * FFD);
  k_cvt_bf16<<<4096, 256, 0, stream>>>(W2,  w2_b,  (long)LL * FFD * EE);
  k_cvt_bf16<<<4096, 256, 0, stream>>>(Wlm, wlm_b, (long)EE * VV);

  // embeddings
  k_embed<<<dim3(EE / 256, MROWS), 256, 0, stream>>>(idx, tok_emb, pos_emb, x);

  for (int l = 0; l < LL; ++l) {
    const bf16* wq_l = wq_b + (size_t)l * EE * EE;
    const bf16* wk_l = wk_b + (size_t)l * EE * EE;
    const bf16* wv_l = wv_b + (size_t)l * EE * EE;
    const bf16* wo_l = wo_b + (size_t)l * EE * EE;
    const bf16* w1_l = w1_b + (size_t)l * EE * FFD;
    const bf16* w2_l = w2_b + (size_t)l * FFD * EE;
    const float* bo_l = bo + (size_t)l * EE;
    const float* b1_l = b1 + (size_t)l * FFD;
    const float* b2_l = b2 + (size_t)l * EE;

    k_ln<<<MROWS, 256, 0, stream>>>(x, ln1s + (size_t)l * EE, ln1b + (size_t)l * EE, hbuf);

    dim3 gEE(EE / 128, MROWS / 128);
    k_gemm<0><<<gEE, 256, 0, stream>>>(hbuf, wq_l, nullptr, nullptr, nullptr, qbuf, EE, EE);
    k_gemm<0><<<gEE, 256, 0, stream>>>(hbuf, wk_l, nullptr, nullptr, nullptr, kbuf, EE, EE);
    k_gemm<0><<<gEE, 256, 0, stream>>>(hbuf, wv_l, nullptr, nullptr, nullptr, vbuf, EE, EE);

    k_attn<<<dim3(TT / 128, BB * HH), 256, 0, stream>>>(qbuf, kbuf, vbuf, obuf);

    k_gemm<2><<<gEE, 256, 0, stream>>>(obuf, wo_l, bo_l, x, x, nullptr, EE, EE);

    k_ln<<<MROWS, 256, 0, stream>>>(x, ln2s + (size_t)l * EE, ln2b + (size_t)l * EE, hbuf);

    k_gemm<1><<<dim3(FFD / 128, MROWS / 128), 256, 0, stream>>>(
        hbuf, w1_l, b1_l, nullptr, nullptr, abuf, FFD, EE);
    k_gemm<2><<<gEE, 256, 0, stream>>>(abuf, w2_l, b2_l, x, x, nullptr, EE, FFD);
  }

  k_ln<<<MROWS, 256, 0, stream>>>(x, lnfs, lnfb, xfbuf);
  k_gemm<3><<<dim3(VV / 128, MROWS / 128), 256, 0, stream>>>(
      xfbuf, wlm_b, blm, nullptr, logits, nullptr, VV, EE);

  k_loss_rows<<<MROWS, 256, 0, stream>>>(logits, targets, rnll);
  k_loss_reduce<<<1, 256, 0, stream>>>(rnll, loss);
}
